// KGGCNRecommender_32349693673727
// MI455X (gfx1250) — compile-verified, matrix operand order checked by reading, not run on hardware
//
#include <hip/hip_runtime.h>
#include <hip/hip_bf16.h>

#define N_NODES 100000
#define N_USERS 50000
#define F_DIM   256
#define HDIM    128
#define N_EDGES 800000
#define BATCH   4096
#define NH      (N_NODES * HDIM)   // 12,800,000 floats = 51.2 MB

typedef __attribute__((ext_vector_type(16))) __bf16 v16bf;
typedef __attribute__((ext_vector_type(8)))  __bf16 v8bf;
typedef __attribute__((ext_vector_type(8)))  float  v8f;

// sum across the 16-lane half of a wave32 (xor masks <16 stay within each half)
__device__ __forceinline__ float half_reduce(float s) {
    s += __shfl_xor(s, 1, 32);
    s += __shfl_xor(s, 2, 32);
    s += __shfl_xor(s, 4, 32);
    s += __shfl_xor(s, 8, 32);
    return s;
}

// ---------------------------------------------------------------------------
// GEMM: Out[r, 0:128] = act( A[r, 0:K] @ W[0:128, 0:K]^T + bias ), optional
// fused ReLU+LayerNorm. One wave32 -> one 16-row x 128-col tile via 8 bf16
// WMMA accumulators. W staged in LDS as bf16 (reused by all 8 waves).
// ---------------------------------------------------------------------------
template <int K, bool LN>
__global__ void __launch_bounds__(256)
gemm_bf16_kernel(const float* __restrict__ A, const float* __restrict__ W,
                 const float* __restrict__ bias, const float* __restrict__ ln_g,
                 const float* __restrict__ ln_b, float* __restrict__ Out,
                 int nRows) {
    __shared__ __attribute__((aligned(16))) __bf16 sW[HDIM * K];  // <= 64 KB

    const int tid  = threadIdx.x;
    // cooperative load + f32->bf16 convert of the whole weight matrix
    for (int i = tid; i < HDIM * K; i += 256) sW[i] = (__bf16)W[i];
    __syncthreads();

    const int wave   = tid >> 5;
    const int lane   = tid & 31;
    const int lane15 = lane & 15;
    const int half   = lane >> 4;          // 0: K 0-7/16-23, 1: K 8-15/24-31
    const int rowBase = blockIdx.x * 128 + wave * 16;
    if (rowBase >= nRows) return;          // whole-wave exit (EXEC stays all-1)

    const float* arow = A + (size_t)(rowBase + lane15) * K;
    const int koff = half * 8;

    v8f acc[8];
#pragma unroll
    for (int t = 0; t < 8; ++t) acc[t] = (v8f)0.0f;

    for (int kc = 0; kc < K; kc += 32) {
        const int kb = kc + koff;
        if (kc + 32 < K) __builtin_prefetch(arow + kb + 32, 0, 1);
        // A fragment: rows across lanes, K packed 2-per-VGPR (16-bit A layout)
        float4 a0 = *(const float4*)(arow + kb);
        float4 a1 = *(const float4*)(arow + kb + 4);
        float4 a2 = *(const float4*)(arow + kb + 16);
        float4 a3 = *(const float4*)(arow + kb + 20);
        v16bf av;
        av[0] = (__bf16)a0.x; av[1] = (__bf16)a0.y; av[2] = (__bf16)a0.z; av[3] = (__bf16)a0.w;
        av[4] = (__bf16)a1.x; av[5] = (__bf16)a1.y; av[6] = (__bf16)a1.z; av[7] = (__bf16)a1.w;
        av[8]  = (__bf16)a2.x; av[9]  = (__bf16)a2.y; av[10] = (__bf16)a2.z; av[11] = (__bf16)a2.w;
        av[12] = (__bf16)a3.x; av[13] = (__bf16)a3.y; av[14] = (__bf16)a3.z; av[15] = (__bf16)a3.w;

#pragma unroll
        for (int t = 0; t < 8; ++t) {
            // B fragment: column (= W row) across lanes, same K striping as A
            const __bf16* wrow = sW + (size_t)(16 * t + lane15) * K;
            v8bf blo = *(const v8bf*)(wrow + kb);
            v8bf bhi = *(const v8bf*)(wrow + kb + 16);
            v16bf bv;
#pragma unroll
            for (int j = 0; j < 8; ++j) { bv[j] = blo[j]; bv[8 + j] = bhi[j]; }
            acc[t] = __builtin_amdgcn_wmma_f32_16x16x32_bf16(
                false, av, false, bv, (short)0, acc[t], false, false);
        }
    }

    // preload per-lane epilogue params (column = 16*t + lane15)
    float bias_r[8], g_r[8], b_r[8];
#pragma unroll
    for (int t = 0; t < 8; ++t) {
        const int col = 16 * t + lane15;
        bias_r[t] = bias[col];
        if (LN) { g_r[t] = ln_g[col]; b_r[t] = ln_b[col]; }
    }

#pragma unroll
    for (int e = 0; e < 8; ++e) {
        const int row = rowBase + e + half * 8;   // C layout: lanes>=16 -> M+8
        float* orow = Out + (size_t)row * HDIM;
        if (LN) {
            float v[8];
            float s = 0.0f;
#pragma unroll
            for (int t = 0; t < 8; ++t) {
                float x = fmaxf(acc[t][e] + bias_r[t], 0.0f);   // bias + ReLU
                v[t] = x; s += x;
            }
            s = half_reduce(s);                    // row sum over all 128 cols
            const float mu = s * (1.0f / HDIM);
            float q = 0.0f;
#pragma unroll
            for (int t = 0; t < 8; ++t) { float d = v[t] - mu; q += d * d; }
            q = half_reduce(q);
            const float rs = rsqrtf(q * (1.0f / HDIM) + 1e-5f);
#pragma unroll
            for (int t = 0; t < 8; ++t)
                orow[16 * t + lane15] = (v[t] - mu) * rs * g_r[t] + b_r[t];
        } else {
#pragma unroll
            for (int t = 0; t < 8; ++t)
                orow[16 * t + lane15] = acc[t][e] + bias_r[t];
        }
    }
}

// ---------------------------------------------------------------------------
// COO SpMM: out[row] += val * x[col]; one wave32 per edge, 4 floats per lane,
// hardware f32 atomics (L2-resident: each 51.2 MB operand fits in 192 MB L2).
// ---------------------------------------------------------------------------
__global__ void __launch_bounds__(256)
spmm_kernel(const int* __restrict__ erow, const int* __restrict__ ecol,
            const float* __restrict__ evals, const float* __restrict__ x,
            float* __restrict__ out, int nEdges) {
    const int e    = (blockIdx.x * 256 + threadIdx.x) >> 5;
    const int lane = threadIdx.x & 31;
    if (e >= nEdges) return;
    const int r = erow[e];
    const int c = ecol[e];
    const float v = evals[e];
    const float4 xv = ((const float4*)(x + (size_t)c * HDIM))[lane];
    float* o = out + (size_t)r * HDIM + lane * 4;
    unsafeAtomicAdd(o + 0, xv.x * v);
    unsafeAtomicAdd(o + 1, xv.y * v);
    unsafeAtomicAdd(o + 2, xv.z * v);
    unsafeAtomicAdd(o + 3, xv.w * v);
}

__global__ void __launch_bounds__(256)
zero_kernel(float4* __restrict__ p, int n4) {
    const int i = blockIdx.x * 256 + threadIdx.x;
    if (i < n4) p[i] = make_float4(0.f, 0.f, 0.f, 0.f);
}

// ---------------------------------------------------------------------------
// Scoring: fuse item_emb = h2 + residual + item_id_emb at the 4096 gathered
// rows only; wave32 per sample, full-wave shfl reduction for the dot product.
// ---------------------------------------------------------------------------
__global__ void __launch_bounds__(256)
score_kernel(const int* __restrict__ uidx, const int* __restrict__ iidx,
             const float* __restrict__ uemb, const float* __restrict__ h2,
             const float* __restrict__ h0, const float* __restrict__ iid,
             const float* __restrict__ ubias, const float* __restrict__ ibias,
             const float* __restrict__ gbias, float* __restrict__ out, int nB) {
    const int b    = (blockIdx.x * 256 + threadIdx.x) >> 5;
    const int lane = threadIdx.x & 31;
    if (b >= nB) return;
    const int u = uidx[b];
    const int it = iidx[b];
    const float4 uv = ((const float4*)(uemb + (size_t)u * HDIM))[lane];
    const float4 av = ((const float4*)(h2 + (size_t)it * HDIM))[lane];
    const float4 bv = ((const float4*)(h0 + (size_t)it * HDIM))[lane];
    const float4 cv = ((const float4*)(iid + (size_t)it * HDIM))[lane];
    float s = uv.x * (av.x + bv.x + cv.x) + uv.y * (av.y + bv.y + cv.y) +
              uv.z * (av.z + bv.z + cv.z) + uv.w * (av.w + bv.w + cv.w);
    s += __shfl_xor(s, 16, 32);
    s += __shfl_xor(s, 8, 32);
    s += __shfl_xor(s, 4, 32);
    s += __shfl_xor(s, 2, 32);
    s += __shfl_xor(s, 1, 32);
    if (lane == 0) out[b] = gbias[0] + ubias[u] + ibias[it] + s;
}

extern "C" void kernel_launch(void* const* d_in, const int* in_sizes, int n_in,
                              void* d_out, int out_size, void* d_ws, size_t ws_size,
                              hipStream_t stream) {
    const float* X        = (const float*)d_in[0];
    const int*   adj_row  = (const int*)d_in[1];
    const int*   adj_col  = (const int*)d_in[2];
    const float* adj_vals = (const float*)d_in[3];
    const int*   user_idx = (const int*)d_in[4];
    const int*   item_idx = (const int*)d_in[5];
    const float* Wp   = (const float*)d_in[6];
    const float* bp   = (const float*)d_in[7];
    const float* W1   = (const float*)d_in[8];
    const float* b1   = (const float*)d_in[9];
    const float* W2   = (const float*)d_in[10];
    const float* b2   = (const float*)d_in[11];
    const float* ln_g = (const float*)d_in[12];
    const float* ln_b = (const float*)d_in[13];
    const float* uemb = (const float*)d_in[14];
    const float* iid  = (const float*)d_in[15];
    const float* ub   = (const float*)d_in[16];
    const float* ib   = (const float*)d_in[17];
    const float* gb   = (const float*)d_in[18];
    float* out = (float*)d_out;

    float* ws   = (float*)d_ws;
    float* h0   = ws;             // residual + layer-1 input   (51.2 MB)
    float* bufA = ws + (size_t)NH;       // layer-1 spmm/mlp    (51.2 MB)
    float* bufB = ws + (size_t)2 * NH;   // layer-2 spmm/mlp    (51.2 MB)

    const int gemmBlocks = (N_NODES + 127) / 128;       // 782
    const int spmmBlocks = (N_EDGES + 7) / 8;           // 100000
    const int zeroBlocks = (NH / 4 + 255) / 256;        // 12500
    const int scoreBlocks = (BATCH + 7) / 8;            // 512

    // h0 = X @ Wp^T + bp
    gemm_bf16_kernel<F_DIM, false><<<gemmBlocks, 256, 0, stream>>>(
        X, Wp, bp, nullptr, nullptr, h0, N_NODES);

    // layer 1: t = adj @ h0 ; h1 = LN(relu(t @ W1^T + b1)) (in-place on bufA)
    zero_kernel<<<zeroBlocks, 256, 0, stream>>>((float4*)bufA, NH / 4);
    spmm_kernel<<<spmmBlocks, 256, 0, stream>>>(adj_row, adj_col, adj_vals,
                                                h0, bufA, N_EDGES);
    gemm_bf16_kernel<HDIM, true><<<gemmBlocks, 256, 0, stream>>>(
        bufA, W1, b1, ln_g, ln_b, bufA, N_NODES);

    // layer 2
    zero_kernel<<<zeroBlocks, 256, 0, stream>>>((float4*)bufB, NH / 4);
    spmm_kernel<<<spmmBlocks, 256, 0, stream>>>(adj_row, adj_col, adj_vals,
                                                bufA, bufB, N_EDGES);
    gemm_bf16_kernel<HDIM, true><<<gemmBlocks, 256, 0, stream>>>(
        bufB, W2, b2, ln_g, ln_b, bufB, N_NODES);

    // fused scoring (item_emb assembled only at the 4096 gathered rows)
    score_kernel<<<scoreBlocks, 256, 0, stream>>>(
        user_idx, item_idx, uemb, bufB, h0, iid, ub, ib, gb, out, BATCH);
}
